// ContextEncodingModule_84464826843236
// MI455X (gfx1250) — matrix operand order repaired and checked by reading, not compiled, not run on hardware
//
#include <hip/hip_runtime.h>
#include <hip/hip_bf16.h>

typedef __attribute__((ext_vector_type(16))) _Float16 v16h;
typedef __attribute__((ext_vector_type(8)))  _Float16 v8h;
typedef __attribute__((ext_vector_type(4)))  _Float16 v4h;
typedef __attribute__((ext_vector_type(8)))  float    v8f;

union F16x16 { v16h v; v8h h[2]; };

#define B_   8
#define N_   4096
#define C_   512
#define K_   32
#define CLS_ 21
#define TN   64      // pixels per tile block
#define PX   520     // LDS pitch for rows over C (512 + 8 pad)
#define PT   72      // LDS pitch for rows over TN (64 + 8 pad)
#define P36  36      // logits pitch (32 + 4 pad)

// workspace layout (float offsets)
#define WS_ENC   0            // [B][K][C] accumulator = 131072
#define WS_WSUM  131072       // [B][K]                = 256
#define WS_ATTN  131328       // [B][C]                = 4096

// ---- A fragment: lane (idx,half) holds row elems kk = {8h..8h+7, 16+8h..16+8h+7}
__device__ __forceinline__ v16h frag_a(const _Float16* row, int base, int half) {
  F16x16 u;
  u.h[0] = *(const v8h*)(row + base + 8 * half);
  u.h[1] = *(const v8h*)(row + base + 16 + 8 * half);
  return u.v;
}
// ---- B fragment: lane (idx,half) holds column elems kk = {8h..8h+7, 16+8h..16+8h+7}
__device__ __forceinline__ v16h frag_b(const _Float16* row, int base, int half) {
  F16x16 u;
  u.h[0] = *(const v8h*)(row + base + 8 * half);
  u.h[1] = *(const v8h*)(row + base + 16 + 8 * half);
  return u.v;
}

__global__ void zero_ws_kernel(float* __restrict__ ws) {
  for (int i = blockIdx.x * blockDim.x + threadIdx.x; i < WS_ATTN;
       i += gridDim.x * blockDim.x)
    ws[i] = 0.f;
}

// One block = one batch's 64-pixel tile. 256 threads = 8 waves.
__global__ __launch_bounds__(256) void encode_tile_kernel(
    const float* __restrict__ x, const float* __restrict__ cw,
    const float* __restrict__ sf, float* __restrict__ ws)
{
  __shared__ _Float16 cwT[K_ * PX];   // codewords^T  (k-major over c)   ~33 KB
  __shared__ _Float16 xn [TN * PX];   // x tile, n-major over c (f16)    ~65 KB
  __shared__ _Float16 xT [C_ * PT];   // x tile, c-major over n (f16)    ~72 KB
  __shared__ _Float16 wTl[K_ * PT];   // softmax weights^T (k-major)     ~4.5 KB
  __shared__ float    logits[TN * P36];                                 // ~9 KB
  __shared__ float    xsq_s[TN];
  __shared__ float    cwsq_s[K_];
  __shared__ float    sf_s[K_];

  const int t    = threadIdx.x;
  const int bb   = blockIdx.x >> 6;   // batch
  const int tile = blockIdx.x & 63;   // 64 tiles per batch
  const int n0   = tile * TN;

  if (t < TN) xsq_s[t] = 0.f;
  __syncthreads();

  // ---------- stage x tile (both layouts) + per-pixel |x|^2 ----------
  {
    const int n  = t >> 2;            // 4 threads per pixel row
    const int cq = t & 3;
    const float* xrow = x + ((size_t)(bb * N_ + n0 + n)) * C_ + cq * 128;
    float part = 0.f;
    #pragma unroll 4
    for (int i = 0; i < 32; ++i) {
      float4 v = ((const float4*)xrow)[i];
      int c = cq * 128 + i * 4;
      v4h h;
      h.x = (_Float16)v.x; h.y = (_Float16)v.y;
      h.z = (_Float16)v.z; h.w = (_Float16)v.w;
      *(v4h*)&xn[n * PX + c] = h;
      xT[(c + 0) * PT + n] = h.x;
      xT[(c + 1) * PT + n] = h.y;
      xT[(c + 2) * PT + n] = h.z;
      xT[(c + 3) * PT + n] = h.w;
      part += v.x * v.x + v.y * v.y + v.z * v.z + v.w * v.w;
    }
    atomicAdd(&xsq_s[n], part);
  }
  // ---------- stage codewords^T, |cw_k|^2, smoothing factors ----------
  for (int i = t; i < C_ * K_; i += 256) {
    int c = i >> 5, k = i & 31;
    cwT[k * PX + c] = (_Float16)cw[i];
  }
  if (t < K_) {
    float s = 0.f;
    for (int c = 0; c < C_; ++c) { float v = cw[c * K_ + t]; s += v * v; }
    cwsq_s[t] = s;
    sf_s[t]   = sf[t];
  }
  __syncthreads();

  const int wv = t >> 5, lane = t & 31, idx = lane & 15, half = lane >> 4;

  // ---------- GEMM1: cross^T[k,n] = CW^T(KxC) @ X^T(CxN), Kdim = C ----------
  {
    const int mt = wv & 1;            // k-tile (2)
    const int nt = wv >> 1;           // n-tile (4)
    const _Float16* arow = &cwT[(mt * 16 + idx) * PX];
    const _Float16* brow = &xn [(nt * 16 + idx) * PX];
    v8f acc = {};
    #pragma unroll
    for (int kc = 0; kc < 16; ++kc) {
      v16h a = frag_a(arow, kc * 32, half);
      v16h b = frag_b(brow, kc * 32, half);
      acc = __builtin_amdgcn_wmma_f32_16x16x32_f16(false, a, false, b,
                                                   (short)0, acc, false, false);
    }
    const int n  = nt * 16 + idx;
    const float xs = xsq_s[n];
    #pragma unroll
    for (int r = 0; r < 8; ++r) {
      int k = mt * 16 + half * 8 + r;             // C/D layout: M = r + 8*half
      float cr = acc[r];
      logits[n * P36 + k] = -(xs - 2.f * cr + cwsq_s[k]) * sf_s[k];
    }
  }
  __syncthreads();

  // ---------- softmax over K per pixel; weights stored transposed ----------
  if (t < TN) {
    float m = -3.4e38f;
    for (int k = 0; k < K_; ++k) m = fmaxf(m, logits[t * P36 + k]);
    float s = 0.f;
    for (int k = 0; k < K_; ++k) {
      float e = __expf(logits[t * P36 + k] - m);
      logits[t * P36 + k] = e; s += e;
    }
    float inv = 1.f / s;
    for (int k = 0; k < K_; ++k)
      wTl[k * PT + t] = (_Float16)(logits[t * P36 + k] * inv);
  }
  __syncthreads();

  // ---------- per-codeword weight mass -> global accumulator ----------
  if (t < K_) {
    float s = 0.f;
    for (int n = 0; n < TN; ++n) s += (float)wTl[t * PT + n];
    atomicAdd(&ws[WS_WSUM + bb * K_ + t], s);
  }

  // ---------- GEMM2: enc^T[k,c] += W^T(KxN) @ X(NxC), Kdim = N(=64) ----------
  {
    const int mt = wv & 1;
    float* encA = ws + WS_ENC + (size_t)bb * (K_ * C_);
    const _Float16* arow = &wTl[(mt * 16 + idx) * PT];
    v16h a0 = frag_a(arow, 0,  half);
    v16h a1 = frag_a(arow, 32, half);
    #pragma unroll
    for (int j = 0; j < 8; ++j) {
      const int ct = (wv >> 1) * 8 + j;           // 32 c-tiles / 4 wave-cols
      const _Float16* brow = &xT[(ct * 16 + idx) * PT];
      v8f acc = {};
      acc = __builtin_amdgcn_wmma_f32_16x16x32_f16(false, a0, false,
              frag_b(brow, 0, half),  (short)0, acc, false, false);
      acc = __builtin_amdgcn_wmma_f32_16x16x32_f16(false, a1, false,
              frag_b(brow, 32, half), (short)0, acc, false, false);
      const int c = ct * 16 + idx;
      #pragma unroll
      for (int r = 0; r < 8; ++r) {
        int k = mt * 16 + half * 8 + r;
        atomicAdd(&encA[k * C_ + c], acc[r]);
      }
    }
  }
}

// One block per batch: enc -> BN -> ReLU -> sum_K, then attn + se heads.
__global__ __launch_bounds__(256) void finalize_kernel(
    const float* __restrict__ cw,
    const float* __restrict__ gamma, const float* __restrict__ beta,
    const float* __restrict__ mean,  const float* __restrict__ var,
    const float* __restrict__ w_enc, const float* __restrict__ b_enc,
    const float* __restrict__ w_se,  const float* __restrict__ b_se,
    float* __restrict__ ws, float* __restrict__ out)
{
  __shared__ float encvec[C_];
  const int bb = blockIdx.x;
  const int t  = threadIdx.x;
  const float* encA = ws + WS_ENC + (size_t)bb * (K_ * C_);
  const float* wsum = ws + WS_WSUM + bb * K_;

  for (int c = t; c < C_; c += 256) {
    float invs = rsqrtf(var[c] + 1e-3f) * gamma[c];
    float mn = mean[c], bt = beta[c];
    float acc = 0.f;
    #pragma unroll 8
    for (int k = 0; k < K_; ++k) {
      float e = encA[k * C_ + c] - cw[c * K_ + k] * wsum[k];
      float v = (e - mn) * invs + bt;
      acc += fmaxf(v, 0.f);
    }
    encvec[c] = acc;
  }
  __syncthreads();

  float* attn = ws + WS_ATTN + bb * C_;
  for (int d = t; d < C_; d += 256) {
    float s = b_enc[d];
    for (int c = 0; c < C_; ++c) s += encvec[c] * w_enc[c * C_ + d];
    attn[d] = 1.f / (1.f + __expf(-s));
  }
  if (t < CLS_) {
    float s = b_se[t];
    for (int c = 0; c < C_; ++c) s += encvec[c] * w_se[c * CLS_ + t];
    out[(size_t)B_ * N_ * C_ + bb * CLS_ + t] = 1.f / (1.f + __expf(-s));
  }
}

// out[b,h,w,c] = attn[b,c] * x[b,h,w,c]  (float4, one element per thread)
__global__ __launch_bounds__(256) void scale_kernel(
    const float* __restrict__ x, const float* __restrict__ ws,
    float* __restrict__ out)
{
  const size_t i4 = (size_t)blockIdx.x * 256 + threadIdx.x;  // float4 index
  const int c4 = (int)(i4 & 127);        // C_/4 = 128 float4 per pixel
  const int bb = (int)(i4 >> 19);        // N_*C_/4 = 2^19 float4 per batch
  const float* attn = ws + WS_ATTN + bb * C_;
  float4 a = *(const float4*)&attn[c4 * 4];
  float4 v = ((const float4*)x)[i4];
  v.x *= a.x; v.y *= a.y; v.z *= a.z; v.w *= a.w;
  ((float4*)out)[i4] = v;
}

extern "C" void kernel_launch(void* const* d_in, const int* in_sizes, int n_in,
                              void* d_out, int out_size, void* d_ws, size_t ws_size,
                              hipStream_t stream) {
  const float* x     = (const float*)d_in[0];
  const float* cw    = (const float*)d_in[1];
  const float* sf    = (const float*)d_in[2];
  const float* gamma = (const float*)d_in[3];
  const float* beta  = (const float*)d_in[4];
  const float* mean  = (const float*)d_in[5];
  const float* var   = (const float*)d_in[6];
  const float* w_enc = (const float*)d_in[7];
  const float* b_enc = (const float*)d_in[8];
  const float* w_se  = (const float*)d_in[9];
  const float* b_se  = (const float*)d_in[10];
  float* out = (float*)d_out;
  float* ws  = (float*)d_ws;

  zero_ws_kernel<<<128, 256, 0, stream>>>(ws);
  encode_tile_kernel<<<B_ * (N_ / TN), 256, 0, stream>>>(x, cw, sf, ws);
  finalize_kernel<<<B_, 256, 0, stream>>>(cw, gamma, beta, mean, var,
                                          w_enc, b_enc, w_se, b_se, ws, out);
  scale_kernel<<<(B_ * N_ * C_) / (4 * 256), 256, 0, stream>>>(x, ws, out);
}
